// GatedDuoAttention_2061584302676
// MI455X (gfx1250) — matrix operand order repaired and checked by reading, not compiled, not run on hardware
//
#include <hip/hip_runtime.h>
#include <hip/hip_bf16.h>

typedef _Float16 v16h __attribute__((ext_vector_type(16)));
typedef float    v8f  __attribute__((ext_vector_type(8)));

#define B_   2
#define S_   2048
#define H_   1024
#define NH_  16
#define HD_  64
#define RH_  4
#define ROWS (B_ * S_)      // 4096
#define KVW  (RH_ * HD_)    // 256
#define WIN  64             // WS/2
#define MASK_NEG (-1e30f)

// ---------------------------------------------------------------------------
// CDNA5 async memory -> LDS DMA (ASYNCcnt path).  Per CDNA5 ISA
// the instruction offset is added to BOTH the LDS and the global address, so
// one base VGPR (pair) + literal offsets covers a whole row copy.
// ---------------------------------------------------------------------------
#define ASYNC_B128(lds, g, off)                                            \
    asm volatile("global_load_async_to_lds_b128 %0, %1, off offset:" #off  \
                 :: "v"(lds), "v"((unsigned long long)(uintptr_t)(g))      \
                 : "memory")

__device__ __forceinline__ void wait_async0() {
    asm volatile("s_wait_asynccnt 0" ::: "memory");
}
__device__ __forceinline__ unsigned lds_off(const void* p) {
    return (unsigned)(uintptr_t)p;   // low 32 bits of flat LDS address = LDS offset
}

// ---------------------------------------------------------------------------
// f32 -> f16 elementwise convert
// ---------------------------------------------------------------------------
__global__ __launch_bounds__(256)
void cvt_f32_f16(const float* __restrict__ in, _Float16* __restrict__ out, int n) {
    int i = blockIdx.x * 256 + threadIdx.x;
    if (i < n) out[i] = (_Float16)in[i];
}

// ---------------------------------------------------------------------------
// Gate kernel: per row LayerNorm -> dot(Wg) -> sigmoid(clamp) ; reg term
// ---------------------------------------------------------------------------
__global__ __launch_bounds__(256)
void gate_kernel(const float* __restrict__ x, const float* __restrict__ Wg,
                 const float* __restrict__ bg, const float* __restrict__ gamma,
                 const float* __restrict__ beta,
                 float* __restrict__ gates, float* __restrict__ regp) {
    __shared__ float red[256];
    const int row = blockIdx.x;
    const int t = threadIdx.x;
    const float* xr = x + (size_t)row * H_;

    float s = 0.f;
    for (int i = t; i < H_; i += 256) s += xr[i];
    red[t] = s; __syncthreads();
    for (int o = 128; o > 0; o >>= 1) { if (t < o) red[t] += red[t + o]; __syncthreads(); }
    float mu = red[0] * (1.0f / H_); __syncthreads();

    float vs = 0.f;
    for (int i = t; i < H_; i += 256) { float d = xr[i] - mu; vs += d * d; }
    red[t] = vs; __syncthreads();
    for (int o = 128; o > 0; o >>= 1) { if (t < o) red[t] += red[t + o]; __syncthreads(); }
    float rstd = rsqrtf(red[0] * (1.0f / H_) + 1e-5f); __syncthreads();

    float dot = 0.f;
    for (int i = t; i < H_; i += 256) {
        float ln = (xr[i] - mu) * rstd * gamma[i] + beta[i];
        dot += ln * Wg[i];
    }
    red[t] = dot; __syncthreads();
    for (int o = 128; o > 0; o >>= 1) { if (t < o) red[t] += red[t + o]; __syncthreads(); }

    if (t == 0) {
        float z = red[0] + bg[0];
        z = fminf(10.f, fmaxf(-10.f, z));
        float g = 1.f / (1.f + expf(-z));
        gates[row] = g;
        float gs = fminf(1.f - 1e-5f, fmaxf(1e-5f, g));
        float reg = 0.1f * g * (1.f - g)
                  - 0.01f * (g * logf(gs) + (1.f - g) * logf(1.f - gs))
                  + 0.1f * g;
        regp[row] = reg;
    }
}

__global__ __launch_bounds__(256)
void reduce_reg(const float* __restrict__ regp, float* __restrict__ out, int n) {
    __shared__ float red[256];
    int t = threadIdx.x;
    float s = 0.f;
    for (int i = t; i < n; i += 256) s += regp[i];
    red[t] = s; __syncthreads();
    for (int o = 128; o > 0; o >>= 1) { if (t < o) red[t] += red[t + o]; __syncthreads(); }
    if (t == 0) out[0] = red[0] / (float)n;
}

// ---------------------------------------------------------------------------
// Generic WMMA GEMM:  C[M,N] = A[M,K] * W[N,K]^T   (PyTorch Linear layout)
// Block tile 128x64, 4 wave32 waves (32 M-rows, 8 WMMAs/K-step per wave).
// Double-buffered LDS tiles filled by async DMA, overlapped with WMMA.
// ---------------------------------------------------------------------------
#define GM 128
#define GN 64
#define GK 32

__global__ __launch_bounds__(128)
void gemm_xwT(const _Float16* __restrict__ A, const _Float16* __restrict__ W,
              void* __restrict__ Cout, int M, int N, int K, int c_is_f16) {
    __shared__ _Float16 As[2][GM][GK + 8];   // row stride 80B (16B aligned)
    __shared__ _Float16 Ws[2][GN][GK + 8];

    const int t = threadIdx.x;           // 0..127
    const int wave = t >> 5;             // 0..3
    const int lane = t & 31;
    const int laneN = lane & 15;
    const int half = (lane >= 16);
    const int m0 = blockIdx.x * GM;
    const int n0 = blockIdx.y * GN;

    v8f acc[2][4];
#pragma unroll
    for (int mi = 0; mi < 2; mi++)
#pragma unroll
        for (int nt = 0; nt < 4; nt++)
            acc[mi][nt] = (v8f){0.f,0.f,0.f,0.f,0.f,0.f,0.f,0.f};

    auto stage = [&](int ib, int k0) {
        const _Float16* srcA = A + (size_t)(m0 + t) * K + k0;
        const unsigned dstA = lds_off(&As[ib][t][0]);
        ASYNC_B128(dstA, srcA, 0);
        ASYNC_B128(dstA, srcA, 16);
        ASYNC_B128(dstA, srcA, 32);
        ASYNC_B128(dstA, srcA, 48);
        if (t < GN) {
            const _Float16* srcW = W + (size_t)(n0 + t) * K + k0;
            const unsigned dstW = lds_off(&Ws[ib][t][0]);
            ASYNC_B128(dstW, srcW, 0);
            ASYNC_B128(dstW, srcW, 16);
            ASYNC_B128(dstW, srcW, 32);
            ASYNC_B128(dstW, srcW, 48);
        }
    };

    stage(0, 0);                         // prologue DMA
    int ib = 0;
    for (int k0 = 0; k0 < K; k0 += GK, ib ^= 1) {
        wait_async0();                   // this wave's DMA for buf[ib] done
        __syncthreads();                 // every wave's DMA for buf[ib] done
        if (k0 + GK < K) stage(ib ^ 1, k0 + GK);   // DMA next tile during math

        // A fragments: two 16x32 tiles (rows wave*32 .. wave*32+31)
        const int kbA = half ? 8 : 0;
        v16h af[2];
#pragma unroll
        for (int mi = 0; mi < 2; mi++) {
            const int arow = wave * 32 + mi * 16 + laneN;
#pragma unroll
            for (int e = 0; e < 8; e++) af[mi][e]     = As[ib][arow][kbA + e];
#pragma unroll
            for (int e = 0; e < 8; e++) af[mi][8 + e] = As[ib][arow][16 + kbA + e];
        }

        const int kbB = half ? 8 : 0;
#pragma unroll
        for (int nt = 0; nt < 4; nt++) {
            const int wrow = nt * 16 + laneN;   // n index
            v16h bf;
#pragma unroll
            for (int e = 0; e < 16; e++) bf[e] = Ws[ib][wrow][kbB + e + (e >= 8 ? 8 : 0)];
#pragma unroll
            for (int mi = 0; mi < 2; mi++)
                acc[mi][nt] = __builtin_amdgcn_wmma_f32_16x16x32_f16(
                    false, af[mi], false, bf, (short)0, acc[mi][nt], false, false);
        }
        __syncthreads();                 // all reads of buf[ib] complete
    }

    const int rbase = half ? 8 : 0;
#pragma unroll
    for (int mi = 0; mi < 2; mi++)
#pragma unroll
        for (int nt = 0; nt < 4; nt++)
#pragma unroll
            for (int r = 0; r < 8; r++) {
                int row = m0 + wave * 32 + mi * 16 + rbase + r;
                int col = n0 + nt * 16 + laneN;
                float v = acc[mi][nt][r];
                if (c_is_f16) ((_Float16*)Cout)[(size_t)row * N + col] = (_Float16)v;
                else          ((float*)Cout)[(size_t)row * N + col] = v;
            }
}

// ---------------------------------------------------------------------------
// Flash-style gated attention.  One wave32 per (b, h, 16-query tile).
// V blocks DMA'd to LDS (overlapped with Q.K^T WMMAs); P bounced through LDS.
// ---------------------------------------------------------------------------
__global__ __launch_bounds__(128)
void attn_kernel(const _Float16* __restrict__ Q, const _Float16* __restrict__ Kb,
                 const _Float16* __restrict__ Vb, const float* __restrict__ gates,
                 _Float16* __restrict__ attn) {
    __shared__ _Float16 Pl[4][16][40];   // per-wave P staging (16 q x 32 keys)
    __shared__ _Float16 Vl[4][32][72];   // per-wave V block (32 keys x 64 d), 144B rows

    const int t = threadIdx.x;
    const int wave = t >> 5;
    const int lane = t & 31;
    const int laneN = lane & 15;
    const int half = (lane >= 16);

    const int gtile = blockIdx.x * 4 + wave;       // 0..4095
    const int qt = gtile & (S_ / 16 - 1);
    const int h  = (gtile >> 7) & (NH_ - 1);
    const int b  = gtile >> 11;
    const int hk = h & (RH_ - 1);                  // kv head = h % RH
    const int q0 = qt * 16;

    // ---- Q fragments (d 0..31 and 32..63) from global ----
    const _Float16* qrow = Q + ((size_t)(b * S_ + q0 + laneN) * H_ + h * HD_);
    const int kbA = half ? 8 : 0;
    v16h aQ[2];
#pragma unroll
    for (int d0 = 0; d0 < 2; d0++) {
#pragma unroll
        for (int e = 0; e < 8; e++)  aQ[d0][e]     = qrow[d0 * 32 + kbA + e];
#pragma unroll
        for (int e = 0; e < 8; e++)  aQ[d0][8 + e] = qrow[d0 * 32 + 16 + kbA + e];
    }

    // ---- per-row gate flags ----
    bool loc[8];
#pragma unroll
    for (int r = 0; r < 8; r++)
        loc[r] = gates[b * S_ + q0 + half * 8 + r] <= 0.5f;
    bool lg = gates[b * S_ + q0 + laneN] <= 0.5f;
    const bool all_local  = __all(lg);
    const bool all_causal = __all(!lg);

    int kstart = 0;
    if (all_local) { kstart = q0 - WIN; if (kstart < 0) kstart = 0; kstart &= ~31; }
    int kend = all_causal ? (q0 + 16) : (q0 + 16 + WIN);
    if (kend > S_) kend = S_;

    float m[8], l[8];
    v8f o[4];
#pragma unroll
    for (int r = 0; r < 8; r++) { m[r] = MASK_NEG; l[r] = 0.f; }
#pragma unroll
    for (int i = 0; i < 4; i++) o[i] = (v8f){0.f,0.f,0.f,0.f,0.f,0.f,0.f,0.f};

    const int kbB = half ? 8 : 0;
    const size_t kvbase = (size_t)b * S_;
    const unsigned vdst = lds_off(&Vl[wave][lane][0]);

    for (int kb = kstart; kb < kend; kb += 32) {
        // ---- async DMA this wave's 32x64 V block into LDS (lane = key row) ----
        {
            const _Float16* vrow = Vb + ((kvbase + kb + lane) * KVW + hk * HD_);
            ASYNC_B128(vdst, vrow, 0);
            ASYNC_B128(vdst, vrow, 16);
            ASYNC_B128(vdst, vrow, 32);
            ASYNC_B128(vdst, vrow, 48);
            ASYNC_B128(vdst, vrow, 64);
            ASYNC_B128(vdst, vrow, 80);
            ASYNC_B128(vdst, vrow, 96);
            ASYNC_B128(vdst, vrow, 112);
        }

        // ---- scores: two 16x16 tiles over d = 0..63 (K direct from global) ----
        v8f s01[2];
#pragma unroll
        for (int st = 0; st < 2; st++) {
            v8f sc = (v8f){0.f,0.f,0.f,0.f,0.f,0.f,0.f,0.f};
            const int key = kb + st * 16 + laneN;
            const _Float16* krow = Kb + ((kvbase + key) * KVW + hk * HD_);
#pragma unroll
            for (int d0 = 0; d0 < 2; d0++) {
                v16h bf;
#pragma unroll
                for (int e = 0; e < 16; e++) bf[e] = krow[d0 * 32 + kbB + e + (e >= 8 ? 8 : 0)];
                sc = __builtin_amdgcn_wmma_f32_16x16x32_f16(
                    false, aQ[d0], false, bf, (short)0, sc, false, false);
            }
            s01[st] = sc;
        }

        // ---- mask + online softmax ----
        float p0[8], p1[8], mn[8], scl[8];
#pragma unroll
        for (int r = 0; r < 8; r++) {
            const int i = q0 + half * 8 + r;
            const int j0 = kb + laneN;
            const int j1 = kb + 16 + laneN;
            bool v0 = loc[r] ? (abs(i - j0) <= WIN) : (j0 <= i);
            bool v1 = loc[r] ? (abs(i - j1) <= WIN) : (j1 <= i);
            float a0 = v0 ? s01[0][r] * 0.125f : MASK_NEG;
            float a1 = v1 ? s01[1][r] * 0.125f : MASK_NEG;
            p0[r] = a0; p1[r] = a1;
            float rm = fmaxf(a0, a1);
#pragma unroll
            for (int off = 1; off < 16; off <<= 1)
                rm = fmaxf(rm, __shfl_xor(rm, off, 32));
            mn[r] = fmaxf(m[r], rm);
            scl[r] = __expf(m[r] - mn[r]);
        }
#pragma unroll
        for (int r = 0; r < 8; r++) {
            p0[r] = (p0[r] > -1e29f) ? __expf(p0[r] - mn[r]) : 0.f;
            p1[r] = (p1[r] > -1e29f) ? __expf(p1[r] - mn[r]) : 0.f;
            float ls = p0[r] + p1[r];
#pragma unroll
            for (int off = 1; off < 16; off <<= 1)
                ls += __shfl_xor(ls, off, 32);
            l[r] = l[r] * scl[r] + ls;
            m[r] = mn[r];
#pragma unroll
            for (int nt = 0; nt < 4; nt++) o[nt][r] *= scl[r];
        }

        // ---- stage P (16x32 f16) and repack as A fragment ----
#pragma unroll
        for (int r = 0; r < 8; r++) {
            Pl[wave][half * 8 + r][laneN]      = (_Float16)p0[r];
            Pl[wave][half * 8 + r][16 + laneN] = (_Float16)p1[r];
        }
        asm volatile("s_wait_dscnt 0" ::: "memory");
        v16h aP;
#pragma unroll
        for (int e = 0; e < 8; e++)  aP[e]     = Pl[wave][laneN][kbA + e];
#pragma unroll
        for (int e = 0; e < 8; e++)  aP[8 + e] = Pl[wave][laneN][16 + kbA + e];

        // ---- o += P @ V : V B-fragments gathered from LDS (post-DMA) ----
        wait_async0();
#pragma unroll
        for (int nt = 0; nt < 4; nt++) {
            const int d = nt * 16 + laneN;
            v16h bf;
#pragma unroll
            for (int e = 0; e < 16; e++) bf[e] = Vl[wave][kbB + e + (e >= 8 ? 8 : 0)][d];
            o[nt] = __builtin_amdgcn_wmma_f32_16x16x32_f16(
                false, aP, false, bf, (short)0, o[nt], false, false);
        }
    }

    // ---- epilogue: normalize, write attn[row, h*64 + col] as f16 ----
#pragma unroll
    for (int r = 0; r < 8; r++) {
        const int row = q0 + half * 8 + r;
        const float inv = (l[r] > 0.f) ? (1.f / l[r]) : 0.f;
#pragma unroll
        for (int nt = 0; nt < 4; nt++) {
            attn[(size_t)(b * S_ + row) * H_ + h * HD_ + nt * 16 + laneN] =
                (_Float16)(o[nt][r] * inv);
        }
    }
}

// ---------------------------------------------------------------------------
// host launch
// ---------------------------------------------------------------------------
extern "C" void kernel_launch(void* const* d_in, const int* in_sizes, int n_in,
                              void* d_out, int out_size, void* d_ws, size_t ws_size,
                              hipStream_t stream) {
    const float* x     = (const float*)d_in[0];
    const float* Wq    = (const float*)d_in[1];
    const float* Wk    = (const float*)d_in[2];
    const float* Wv    = (const float*)d_in[3];
    const float* Wo    = (const float*)d_in[4];
    const float* Wg    = (const float*)d_in[5];
    const float* bg    = (const float*)d_in[6];
    const float* gamma = (const float*)d_in[7];
    const float* beta  = (const float*)d_in[8];

    char* ws = (char*)d_ws;
    size_t off = 0;
    auto carve = [&](size_t bytes) { char* p = ws + off; off += (bytes + 255) & ~(size_t)255; return p; };
    _Float16* xh   = (_Float16*)carve((size_t)ROWS * H_ * 2);
    _Float16* WqH  = (_Float16*)carve((size_t)H_ * H_ * 2);
    _Float16* WkH  = (_Float16*)carve((size_t)KVW * H_ * 2);
    _Float16* WvH  = (_Float16*)carve((size_t)KVW * H_ * 2);
    _Float16* WoH  = (_Float16*)carve((size_t)H_ * H_ * 2);
    _Float16* Qb   = (_Float16*)carve((size_t)ROWS * H_ * 2);
    _Float16* Kb   = (_Float16*)carve((size_t)ROWS * KVW * 2);
    _Float16* Vb   = (_Float16*)carve((size_t)ROWS * KVW * 2);
    _Float16* Ab   = (_Float16*)carve((size_t)ROWS * H_ * 2);
    float*    gts  = (float*)carve((size_t)ROWS * 4);
    float*    regp = (float*)carve((size_t)ROWS * 4);

    auto cvt = [&](const float* src, _Float16* dst, int n) {
        cvt_f32_f16<<<(n + 255) / 256, 256, 0, stream>>>(src, dst, n);
    };
    cvt(x,  xh,  ROWS * H_);
    cvt(Wq, WqH, H_ * H_);
    cvt(Wk, WkH, KVW * H_);
    cvt(Wv, WvH, KVW * H_);
    cvt(Wo, WoH, H_ * H_);

    gate_kernel<<<ROWS, 256, 0, stream>>>(x, Wg, bg, gamma, beta, gts, regp);

    gemm_xwT<<<dim3(ROWS / GM, H_ / GN),  128, 0, stream>>>(xh, WqH, Qb, ROWS, H_,  H_, 1);
    gemm_xwT<<<dim3(ROWS / GM, KVW / GN), 128, 0, stream>>>(xh, WkH, Kb, ROWS, KVW, H_, 1);
    gemm_xwT<<<dim3(ROWS / GM, KVW / GN), 128, 0, stream>>>(xh, WvH, Vb, ROWS, KVW, H_, 1);

    attn_kernel<<<1024, 128, 0, stream>>>(Qb, Kb, Vb, gts, Ab);

    gemm_xwT<<<dim3(ROWS / GM, H_ / GN), 128, 0, stream>>>(Ab, WoH, d_out, ROWS, H_, H_, 0);

    reduce_reg<<<1, 256, 0, stream>>>(regp, (float*)d_out + (size_t)ROWS * H_, ROWS);
}
